// VectorQuantizer_74792560493036
// MI455X (gfx1250) — matrix operand order repaired and checked by reading, not compiled, not run on hardware
//
#include <hip/hip_runtime.h>
#include <hip/hip_bf16.h>
#include <stddef.h>
#include <stdint.h>

// ---------------------------------------------------------------------------
// Vector quantizer for MI455X (gfx1250, wave32, WMMA).
//   scores(N,K) = X(N,64) . E(K,64)^T via v_wmma_f32_16x16x32_bf16,
//   3-term bf16 split (hi*hi + hi*lo + lo*hi) ~= fp32 accuracy.
//   argmax over k of (x.e - 0.5||e||^2) == argmin squared distance.
// Packed codebook fragments staged in LDS via async global->LDS DMA
// (global_load_async_to_lds_b128, ASYNCcnt), shared by 8 waves; each wave
// runs 2 token tiles per A-fragment fetch (12 WMMAs / code tile).
// ---------------------------------------------------------------------------

typedef __attribute__((ext_vector_type(16))) __bf16 v16bf;
typedef __attribute__((ext_vector_type(8)))  float  v8f;
typedef __attribute__((ext_vector_type(4)))  float  v4f;

#define BATCH   16
#define DIM     64
#define TLEN    8192
#define KCODES  1024
#define NTOK    (BATCH * TLEN)          // 131072
#define NTILES  (NTOK / 16)             // 8192 token tiles
#define CTILES  (KCODES / 16)           // 64 code tiles
#define TILES_PER_BATCH (TLEN / 16)     // 512

// ws/LDS layout (contiguous, bytes):
//   [0,       128K)  A-hi packed bf16 fragments: (ctile, kstep, lane, 16)
//   [128K,    256K)  A-lo packed bf16 fragments
//   [256K,    260K)  -0.5*||e_k||^2, 1024 floats
#define A_FRAGS    (CTILES * 2 * 32)           // 4096 v16bf per region
#define A_BYTES    (A_FRAGS * 32)              // 131072 B per region
#define STAGE_B    (2 * A_BYTES + KCODES * 4)  // 266240 B total
#define STAGE_CH   (STAGE_B / 16)              // 16640 x b128 chunks

__device__ inline __bf16 bf_hi(float x, float& hf) {
    unsigned u = __builtin_bit_cast(unsigned, x);
    unsigned short hs = (unsigned short)(u >> 16);          // truncate to bf16
    hf = __builtin_bit_cast(float, (unsigned)hs << 16);
    return __builtin_bit_cast(__bf16, hs);
}
__device__ inline __bf16 bf_of(float x) {
    unsigned u = __builtin_bit_cast(unsigned, x);
    return __builtin_bit_cast(__bf16, (unsigned short)(u >> 16));
}

// ---------------------------------------------------------------------------
// Prepass 1: pack codebook into WMMA A-fragment lane order, split hi/lo.
// A (16x32 bf16): lane m = row; VGPR v, half h holds
//   k = (v>=4 ? 16:0) + (lane>=16 ? 8:0) + 2*(v&3) + h.
// ---------------------------------------------------------------------------
__global__ void vq_pack_codebook(const float* __restrict__ cb,
                                 __bf16* __restrict__ ahi,
                                 __bf16* __restrict__ alo) {
    int tid   = blockIdx.x * blockDim.x + threadIdx.x;     // 0..4095
    int ctile = tid >> 6;
    int kstep = (tid >> 5) & 1;
    int lane  = tid & 31;
    int m     = lane & 15;
    const float* row = cb + (size_t)(ctile * 16 + m) * DIM;
    __bf16* oh = ahi + (size_t)tid * 16;
    __bf16* ol = alo + (size_t)tid * 16;
#pragma unroll
    for (int j = 0; j < 16; ++j) {
        int v = j >> 1, h = j & 1;
        int k = ((v >= 4) ? 16 : 0) + ((lane >= 16) ? 8 : 0) + 2 * (v & 3) + h;
        float x = row[kstep * 32 + k];
        float hf;
        __bf16 hi = bf_hi(x, hf);
        oh[j] = hi;
        ol[j] = bf_of(x - hf);
    }
}

// Prepass 2: negated half squared norms (C-accumulator bias).
__global__ void vq_esq(const float* __restrict__ cb, float* __restrict__ nesq) {
    int c = blockIdx.x * blockDim.x + threadIdx.x;         // 0..1023
    const v4f* r = (const v4f*)(cb + (size_t)c * DIM);
    float s = 0.f;
#pragma unroll
    for (int g = 0; g < 16; ++g) {
        v4f q = r[g];
        s += q.x * q.x + q.y * q.y + q.z * q.z + q.w * q.w;
    }
    nesq[c] = -0.5f * s;
}

#define WMMA(A, B, C) \
    __builtin_amdgcn_wmma_f32_16x16x32_bf16(false, (A), false, (B), (short)0, (C), false, false)

// Final per-token reduction + output emit for one 16-token tile.
__device__ inline void vq_emit(float best, int bidx, int lane, int b, int t0,
                               const float* __restrict__ cb,
                               float* __restrict__ outq,
                               int* __restrict__ outidx) {
    // combine the two half-columns (C rows 0-7 in lanes 0-15, 8-15 in 16-31)
    float ob = __shfl_xor(best, 16, 32);
    int   oi = __shfl_xor(bidx, 16, 32);
    if (ob > best || (ob == best && oi < bidx)) { best = ob; bidx = oi; }
    int n = lane & 15;
    if (lane < 16) outidx[b * TLEN + t0 + n] = bidx;
    int dbase = (lane >> 4) << 5;             // lanes 0-15: d 0..31; 16-31: 32..63
    const v4f* crow = (const v4f*)(cb + (size_t)bidx * DIM + dbase);
    float* op = outq + ((size_t)b * DIM + dbase) * TLEN + t0 + n;
#pragma unroll
    for (int g = 0; g < 8; ++g) {
        v4f q = crow[g];                      // per-lane contiguous, L2-hot
        op[(size_t)(4 * g + 0) * TLEN] = q.x; // 16 lanes -> 64B coalesced
        op[(size_t)(4 * g + 1) * TLEN] = q.y;
        op[(size_t)(4 * g + 2) * TLEN] = q.z;
        op[(size_t)(4 * g + 3) * TLEN] = q.w;
    }
}

// ---------------------------------------------------------------------------
// Main kernel: 8 waves/block share LDS-staged codebook; each wave owns two
// 16-token tiles and sweeps 64 code tiles with 12 WMMAs each.
// ---------------------------------------------------------------------------
__global__ __launch_bounds__(256)
void vq_main(const float* __restrict__ in,        // (B, D, T)
             const float* __restrict__ cb,        // (K, D)
             const char* __restrict__ wsbase,     // packed AHI|ALO|NESQ
             float* __restrict__ outq,            // (B, D, T)
             int* __restrict__ outidx) {          // (B, T)
    extern __shared__ __align__(16) char smem[]; // STAGE_B bytes
    v16bf* sAH = (v16bf*)smem;
    v16bf* sAL = sAH + A_FRAGS;
    float* sNE = (float*)(sAL + A_FRAGS);

    // ---- Async DMA: ws -> LDS (all 256 threads, b128 chunks) --------------
    {
        unsigned ldsbase = (unsigned)(uintptr_t)smem;      // low 32 = LDS offset
        for (int i = threadIdx.x; i < STAGE_CH; i += 256) {
            unsigned loff = ldsbase + (unsigned)i * 16u;
            uint64_t g    = (uint64_t)(uintptr_t)(wsbase + (size_t)i * 16);
            asm volatile("global_load_async_to_lds_b128 %0, %1, off"
                         :: "v"(loff), "v"(g) : "memory");
        }
        asm volatile("s_wait_asynccnt 0" ::: "memory");
    }

    const int lane = threadIdx.x & 31;
    const int wave = threadIdx.x >> 5;
    const int tile = blockIdx.x * 16 + wave * 2;  // this wave: tile, tile+1
    const int b    = tile / TILES_PER_BATCH;      // blocks never straddle batches
    const int t0   = (tile % TILES_PER_BATCH) * 16;

    // ---- Build B fragments (d x token) for both tiles, hi/lo split --------
    // B (32x16 bf16): lane holds column n = lane&15; element j <->
    // k = (j>=8 ? 16:0) + (lane>=16 ? 8:0) + (j&7) (matches A packing).
    const int n     = lane & 15;
    const int khalf = (lane >> 4) << 4;           // 0 or 16
    v16bf bh[2][2], bl[2][2];                     // [token tile][kstep]
#pragma unroll
    for (int tt = 0; tt < 2; ++tt) {
        const float* src = in + (size_t)b * DIM * TLEN + t0 + tt * 16 + n;
#pragma unroll
        for (int ks = 0; ks < 2; ++ks) {
#pragma unroll
            for (int j = 0; j < 16; ++j) {
                int d = ks * 32 + ((j >> 3) << 4) + (khalf >> 1) + (j & 7);
                float x = src[(size_t)d * TLEN];  // lanes 0-15 coalesce 64B
                float hf;
                bh[tt][ks][j] = bf_hi(x, hf);
                bl[tt][ks][j] = bf_of(x - hf);
            }
        }
    }

    __syncthreads();                              // LDS codebook ready

    // ---- Sweep code tiles -------------------------------------------------
    float best0 = -3.4e38f, best1 = -3.4e38f;
    int   bidx0 = 0, bidx1 = 0;
    const int rbase = (lane >> 4) << 3;           // C rows: m = rbase + r

#pragma unroll 1
    for (int ct = 0; ct < CTILES; ++ct) {
        int fb = (ct * 2) * 32 + lane;
        v16bf ah0 = sAH[fb], ah1 = sAH[fb + 32];  // ds_load_b128 x2 each
        v16bf al0 = sAL[fb], al1 = sAL[fb + 32];

        const v4f* ep = (const v4f*)(sNE + ct * 16 + rbase);
        v4f ea = ep[0], eb = ep[1];
        v8f c0;                                   // bias C with -0.5||e||^2
        c0[0] = ea.x; c0[1] = ea.y; c0[2] = ea.z; c0[3] = ea.w;
        c0[4] = eb.x; c0[5] = eb.y; c0[6] = eb.z; c0[7] = eb.w;
        v8f c1 = c0;

        c0 = WMMA(ah0, bh[0][0], c0);  c1 = WMMA(ah0, bh[1][0], c1);
        c0 = WMMA(ah1, bh[0][1], c0);  c1 = WMMA(ah1, bh[1][1], c1);
        c0 = WMMA(ah0, bl[0][0], c0);  c1 = WMMA(ah0, bl[1][0], c1);
        c0 = WMMA(ah1, bl[0][1], c0);  c1 = WMMA(ah1, bl[1][1], c1);
        c0 = WMMA(al0, bh[0][0], c0);  c1 = WMMA(al0, bh[1][0], c1);
        c0 = WMMA(al1, bh[0][1], c0);  c1 = WMMA(al1, bh[1][1], c1);

#pragma unroll
        for (int r = 0; r < 8; ++r) {
            int m = ct * 16 + rbase + r;          // ascending m => first-min ties
            float s0 = c0[r];
            if (s0 > best0) { best0 = s0; bidx0 = m; }
            float s1 = c1[r];
            if (s1 > best1) { best1 = s1; bidx1 = m; }
        }
    }

    // ---- Emit both tiles --------------------------------------------------
    vq_emit(best0, bidx0, lane, b, t0,      cb, outq, outidx);
    vq_emit(best1, bidx1, lane, b, t0 + 16, cb, outq, outidx);
}

// ---------------------------------------------------------------------------
extern "C" void kernel_launch(void* const* d_in, const int* in_sizes, int n_in,
                              void* d_out, int out_size, void* d_ws, size_t ws_size,
                              hipStream_t stream) {
    const float* in = (const float*)d_in[0];      // (16, 64, 8192) f32
    const float* cb = (const float*)d_in[1];      // (1024, 64) f32

    __bf16* ahi  = (__bf16*)d_ws;
    __bf16* alo  = ahi + A_FRAGS * 16;
    float*  nesq = (float*)((char*)d_ws + 2 * A_BYTES);

    float* outq   = (float*)d_out;                            // 16*64*8192 floats
    int*   outidx = (int*)((float*)d_out + (size_t)BATCH * DIM * TLEN);

    vq_pack_codebook<<<16, 256, 0, stream>>>(cb, ahi, alo);
    vq_esq<<<4, 256, 0, stream>>>(cb, nesq);
    vq_main<<<NTILES / 16, 256, STAGE_B, stream>>>(in, cb, (const char*)d_ws,
                                                   outq, outidx);
}
